// RotarySelfAttention_67207648247931
// MI455X (gfx1250) — matrix operand order repaired and checked by reading, not compiled, not run on hardware
//
#include <hip/hip_runtime.h>
#include <hip/hip_bf16.h>

// ---------------------------------------------------------------------------
// CDNA5 (gfx1250) bf16 WMMA attention pipeline with TDM (tensor_load_to_lds)
// staging of K/V tiles in the flash-attention kernel.
// ---------------------------------------------------------------------------

typedef __attribute__((ext_vector_type(16))) __bf16 v16bf;
typedef __attribute__((ext_vector_type(8)))  float  v8f;
typedef __attribute__((ext_vector_type(4)))  unsigned int u32x4;
typedef __attribute__((ext_vector_type(8)))  int i32x8;
typedef __attribute__((ext_vector_type(4)))  int i32x4;
typedef unsigned short ushort_t;

union ABFrag { v16bf bf; uint4 q[2]; };
union CFrag  { v8f v; float f[8]; };

__device__ __forceinline__ ushort_t f2bf(float f) {
    unsigned int u = __float_as_uint(f);
    u = (u + 0x7FFFu + ((u >> 16) & 1u)) >> 16;   // round-to-nearest-even
    return (ushort_t)u;
}

__device__ __forceinline__ v8f wmma_bf16(const ABFrag& a, const ABFrag& b, v8f c) {
    return __builtin_amdgcn_wmma_f32_16x16x32_bf16(
        false, a.bf, false, b.bf, (short)0, c, false, false);
}

// ---------------------------------------------------------------------------
// TDM: 2D tile load global->LDS, bf16 elements, optional LDS row padding.
// D# packing per CDNA5 ISA sect. 8 (group0/group1 bitfields).
//   pad_interval_code: rows of 2^(code+1) DWORDs between pads
//   pad_amount_code:   (code+1) DWORDs of LDS padding inserted
// ---------------------------------------------------------------------------
__device__ __forceinline__ void tdm_load_2d(unsigned lds_off, const void* gptr,
                                            unsigned tile_d0, unsigned tile_d1,
                                            unsigned long long row_stride_elems,
                                            unsigned pad_interval_code,
                                            unsigned pad_amount_code) {
    unsigned long long ga = (unsigned long long)(size_t)gptr;
    u32x4 g0;
    g0.x = 1u;                                        // count=1, user descriptor
    g0.y = lds_off;                                   // lds_addr (bytes)
    g0.z = (unsigned)(ga & 0xFFFFFFFFu);              // global_addr[31:0]
    g0.w = (unsigned)((ga >> 32) & 0x1FFFFFFu)        // global_addr[56:32]
         | (2u << 30);                                // type = 2 ("image")
    i32x8 g1;
    unsigned w0 = (1u << 16)                          // data_size = 2 bytes
                | (1u << 20)                          // pad_enable
                | (pad_interval_code << 22)
                | (pad_amount_code << 25);
    unsigned td0 = tile_d0, td1 = tile_d1;            // tensor dims = tile dims (no OOB)
    g1[0] = (int)w0;
    g1[1] = (int)((td0 & 0xFFFFu) << 16);             // [63:48] tensor_dim0 lo
    g1[2] = (int)(((td0 >> 16) & 0xFFFFu) | ((td1 & 0xFFFFu) << 16));
    g1[3] = (int)(((td1 >> 16) & 0xFFFFu) | ((tile_d0 & 0xFFFFu) << 16)); // tile_dim0
    g1[4] = (int)(tile_d1 & 0xFFFFu);                 // tile_dim1; tile_dim2=0 (unused)
    g1[5] = (int)(row_stride_elems & 0xFFFFFFFFull);  // tensor_dim0_stride lo
    g1[6] = (int)((row_stride_elems >> 32) & 0xFFFFull);
    g1[7] = 0;
    i32x4 z4 = {0, 0, 0, 0};
#if __has_include(<hip/amd_detail/amd_gfx1250_TDM.h>)
    i32x8 z8 = {0, 0, 0, 0, 0, 0, 0, 0};
    __builtin_amdgcn_tensor_load_to_lds(g0, g1, z4, z4, z8, 0);
#else
    __builtin_amdgcn_tensor_load_to_lds(g0, g1, z4, z4, 0);
#endif
}

// ---------------------------------------------------------------------------
// Kernel 1: fp32 -> bf16 elementwise convert
// ---------------------------------------------------------------------------
__global__ void f2bf_kernel(const float* __restrict__ src, ushort_t* __restrict__ dst,
                            unsigned int n) {
    unsigned int i = blockIdx.x * 256u + threadIdx.x;
    if (i < n) dst[i] = f2bf(src[i]);
}

// ---------------------------------------------------------------------------
// Kernel 2: transpose+convert  src f32 [K][N] -> dst bf16 [N][K]
// ---------------------------------------------------------------------------
__global__ void transpose_f2bf(const float* __restrict__ src, ushort_t* __restrict__ dst,
                               int K, int N) {
    __shared__ float tile[32][33];
    int n = blockIdx.x * 32 + threadIdx.x;
    int k = blockIdx.y * 32 + threadIdx.y;
#pragma unroll
    for (int i = 0; i < 32; i += 8)
        tile[threadIdx.y + i][threadIdx.x] = src[(size_t)(k + i) * N + n];
    __syncthreads();
    int k2 = blockIdx.y * 32 + threadIdx.x;
    int n2 = blockIdx.x * 32 + threadIdx.y;
#pragma unroll
    for (int i = 0; i < 32; i += 8)
        dst[(size_t)(n2 + i) * K + k2] = f2bf(tile[threadIdx.x][threadIdx.y + i]);
}

// ---------------------------------------------------------------------------
// Kernel 3: bf16 WMMA GEMM  C[M][N] = A[M][K] * Bt[N][K]^T + bias
// Block tile 64(M) x 128(N), 8 waves, K staged in LDS chunks of 32.
// NOTE: prefetches are unconditional (branch-free inner loop); the +GKC
// address on the last K-step stays inside the contiguous d_ws allocation.
// ---------------------------------------------------------------------------
#define GBM 64
#define GBN 128
#define GKC 32
#define LPITCH 40

__global__ __launch_bounds__(256)
void gemm_bf16(const ushort_t* __restrict__ A, const ushort_t* __restrict__ Bt,
               const float* __restrict__ bias, float* __restrict__ C,
               int M, int N, int K) {
    __shared__ ushort_t As[GBM * LPITCH];
    __shared__ ushort_t Bs[GBN * LPITCH];

    const int n0   = blockIdx.x * GBN;
    const int m0   = blockIdx.y * GBM;
    const int tid  = threadIdx.x;
    const int lane = tid & 31;
    const int wave = tid >> 5;
    const int mi   = (wave & 3) * 16;
    const int nw   = (wave >> 2) * 64;

    CFrag acc[4];
#pragma unroll
    for (int on = 0; on < 4; ++on)
#pragma unroll
        for (int r = 0; r < 8; ++r) acc[on].f[r] = 0.0f;

    const int row16 = lane & 15;
    const int half  = lane >> 4;
    const int arow = tid >> 2, ach = tid & 3;

    for (int k0 = 0; k0 < K; k0 += GKC) {
        __syncthreads();
        {   // stage A: 64 rows x 32 cols
            const ushort_t* gp = A + (size_t)(m0 + arow) * K + k0 + ach * 8;
            *(uint4*)(As + arow * LPITCH + ach * 8) = *(const uint4*)gp;
            __builtin_prefetch(gp + GKC, 0, 1);
        }
#pragma unroll
        for (int i = 0; i < 2; ++i) {   // stage B: 128 rows x 32 cols
            int idx = tid + i * 256;
            int row = idx >> 2, ch = idx & 3;
            const ushort_t* gp = Bt + (size_t)(n0 + row) * K + k0 + ch * 8;
            *(uint4*)(Bs + row * LPITCH + ch * 8) = *(const uint4*)gp;
            __builtin_prefetch(gp + GKC, 0, 1);
        }
        __syncthreads();

        ABFrag a;
        const ushort_t* pa = As + (mi + row16) * LPITCH + half * 8;
        a.q[0] = *(const uint4*)pa;
        a.q[1] = *(const uint4*)(pa + 16);

        ABFrag b[4];   // preload all B fragments, then run the WMMA chain
#pragma unroll
        for (int on = 0; on < 4; ++on) {
            const ushort_t* pb = Bs + (nw + on * 16 + row16) * LPITCH + half * 8;
            b[on].q[0] = *(const uint4*)pb;
            b[on].q[1] = *(const uint4*)(pb + 16);
        }
#pragma unroll
        for (int on = 0; on < 4; ++on)
            acc[on].v = wmma_bf16(a, b[on], acc[on].v);
    }

    const int rb = half * 8;
#pragma unroll
    for (int on = 0; on < 4; ++on) {
        int n = n0 + nw + on * 16 + row16;
        float bv = bias[n];
#pragma unroll
        for (int r = 0; r < 8; ++r)
            C[(size_t)(m0 + mi + rb + r) * N + n] = acc[on].f[r] + bv;
    }
}

// ---------------------------------------------------------------------------
// Kernel 4: RoPE + repack (faithful to the reference reshape quirk:
// effective position p=(t*H+h) mod T, pairing d <-> d±32 within a head).
// ---------------------------------------------------------------------------
__global__ void rope_pack(const float* __restrict__ qkv,
                          ushort_t* __restrict__ Qb, ushort_t* __restrict__ Kb,
                          ushort_t* __restrict__ Vt) {
    const int T = 2048;
    unsigned int tid = blockIdx.x * 256u + threadIdx.x;
    int d = tid & 63;
    int h = (tid >> 6) & 15;
    int t = (tid >> 10) & 2047;
    int b = tid >> 21;

    const float* row = qkv + (size_t)(b * T + t) * 3072;
    int c0 = h * 64 + d;
    float qx = row[c0], kx = row[1024 + c0], vx = row[2048 + c0];
    int pd = (d < 32) ? (c0 + 32) : (c0 - 32);
    float qy = row[pd], ky = row[1024 + pd];

    int p  = (t * 16 + h) & 2047;
    int jj = d & 31;
    float freq = __expf(-(float)jj * 0.28782313662425575f);  // ln(10000)/32
    float sn, cs;
    __sincosf((float)p * freq, &sn, &cs);

    float qo, ko;
    if (d < 32) { qo = qx * cs - qy * sn; ko = kx * cs - ky * sn; }
    else        { qo = qx * cs + qy * sn; ko = kx * cs + ky * sn; }

    size_t qkidx = ((size_t)((b * 16 + h) * T + t)) * 64 + d;
    Qb[qkidx] = f2bf(qo * 0.125f);   // fold 1/sqrt(hd) into Q (exact pow2)
    Kb[qkidx] = f2bf(ko);
    Vt[((size_t)((b * 16 + h) * 64 + d)) * T + t] = f2bf(vx);
}

// ---------------------------------------------------------------------------
// Kernel 5: flash attention with TDM-staged, double-buffered K/V tiles.
// Grid (T/128, B*H), 256 threads = 8 waves, each wave owns 16 query rows.
// K/V tiles (64x64 bf16) are DMA'd by wave 0 via tensor_load_to_lds with
// row padding to a 72-element LDS pitch; TENSORcnt + workgroup barriers
// order DMA vs compute; DMA of tile j+1 overlaps compute of tile j.
// ---------------------------------------------------------------------------
#define KVP 72   // padded LDS pitch for K/V tiles (bf16 elems)

__global__ __launch_bounds__(256)
void attn_kernel(const ushort_t* __restrict__ Qb, const ushort_t* __restrict__ Kb,
                 const ushort_t* __restrict__ Vt, ushort_t* __restrict__ Ob) {
    const int T = 2048, HD = 64;
    const int bh   = blockIdx.y;
    const int wave = threadIdx.x >> 5;
    const int lane = threadIdx.x & 31;
    const int q0   = blockIdx.x * 128 + wave * 16;

    const ushort_t* Qp = Qb + (size_t)bh * T * HD;
    const ushort_t* Kp = Kb + (size_t)bh * T * HD;
    const ushort_t* Vp = Vt + (size_t)bh * HD * T;

    __shared__ ushort_t Ks[2][64 * KVP];    // 2 x 9 KB
    __shared__ ushort_t Vs[2][64 * KVP];    // 2 x 9 KB
    __shared__ ushort_t Plds[8 * 16 * 64];  // 16 KB per-wave P staging
    ushort_t* myP = Plds + wave * (16 * 64);

    const int col  = lane & 15;
    const int half = lane >> 4;
    const int rb   = half * 8;

    // Q fragments (16 x 64 = two 16x32 A-operands), loaded once from global
    ABFrag qa[2];
    {
        const ushort_t* p = Qp + (size_t)(q0 + col) * HD + half * 8;
        qa[0].q[0] = *(const uint4*)(p);
        qa[0].q[1] = *(const uint4*)(p + 16);
        qa[1].q[0] = *(const uint4*)(p + 32);
        qa[1].q[1] = *(const uint4*)(p + 48);
    }

    CFrag O[4];
    float m_r[8], l_r[8];
#pragma unroll
    for (int on = 0; on < 4; ++on)
#pragma unroll
        for (int r = 0; r < 8; ++r) O[on].f[r] = 0.0f;
#pragma unroll
    for (int r = 0; r < 8; ++r) { m_r[r] = -1e30f; l_r[r] = 0.0f; }

    const int NT = T / 64;   // 32 key tiles

    // prologue DMA: tile 0 into buffer 0 (rows of 32 DWORDs -> pad codes 4,3)
    if (wave == 0) {
        tdm_load_2d((unsigned)(size_t)&Ks[0][0], Kp, 64, 64, 64ull, 4u, 3u);
        tdm_load_2d((unsigned)(size_t)&Vs[0][0], Vp, 64, 64, (unsigned long long)T, 4u, 3u);
    }

    for (int jt = 0; jt < NT; ++jt) {
        const int cur = jt & 1;
        __syncthreads();   // all compute on buffer cur^1 finished -> free for DMA
        if (wave == 0) {
            if (jt + 1 < NT) {
                int j1 = (jt + 1) * 64;
                tdm_load_2d((unsigned)(size_t)&Ks[cur ^ 1][0],
                            Kp + (size_t)j1 * HD, 64, 64, 64ull, 4u, 3u);
                tdm_load_2d((unsigned)(size_t)&Vs[cur ^ 1][0],
                            Vp + j1, 64, 64, (unsigned long long)T, 4u, 3u);
                __builtin_amdgcn_s_wait_tensorcnt(2);   // tile jt done, jt+1 in flight
            } else {
                __builtin_amdgcn_s_wait_tensorcnt(0);
            }
        }
        __syncthreads();   // release: tile jt visible in LDS to all waves
        const ushort_t* Kt = Ks[cur];
        const ushort_t* Vl = Vs[cur];

        // ---- S = Q K^T  (four 16x16 tiles), B-operands from LDS ----
        CFrag S[4];
#pragma unroll
        for (int nn = 0; nn < 4; ++nn) {
            ABFrag b0, b1;
            const ushort_t* pk = Kt + (nn * 16 + col) * KVP + half * 8;
            b0.q[0] = *(const uint4*)(pk);       b0.q[1] = *(const uint4*)(pk + 16);
            b1.q[0] = *(const uint4*)(pk + 32);  b1.q[1] = *(const uint4*)(pk + 48);
            v8f z = {0.f, 0.f, 0.f, 0.f, 0.f, 0.f, 0.f, 0.f};
            z = wmma_bf16(qa[0], b0, z);
            S[nn].v = wmma_bf16(qa[1], b1, z);
        }

        // ---- online softmax (rows striped across 16-lane groups) ----
        float mj[8];
#pragma unroll
        for (int r = 0; r < 8; ++r) {
            float v = fmaxf(fmaxf(S[0].f[r], S[1].f[r]), fmaxf(S[2].f[r], S[3].f[r]));
            v = fmaxf(v, __shfl_xor(v, 1, 32));
            v = fmaxf(v, __shfl_xor(v, 2, 32));
            v = fmaxf(v, __shfl_xor(v, 4, 32));
            v = fmaxf(v, __shfl_xor(v, 8, 32));
            mj[r] = v;
        }
#pragma unroll
        for (int r = 0; r < 8; ++r) {
            float mn    = fmaxf(m_r[r], mj[r]);
            float alpha = __expf(m_r[r] - mn);
            m_r[r] = mn;
            float rs = 0.0f;
#pragma unroll
            for (int nn = 0; nn < 4; ++nn) {
                float pv = __expf(S[nn].f[r] - mn);
                S[nn].f[r] = pv;
                rs += pv;
            }
            rs += __shfl_xor(rs, 1, 32);
            rs += __shfl_xor(rs, 2, 32);
            rs += __shfl_xor(rs, 4, 32);
            rs += __shfl_xor(rs, 8, 32);
            l_r[r] = l_r[r] * alpha + rs;
#pragma unroll
            for (int on = 0; on < 4; ++on) O[on].f[r] *= alpha;
        }

        // ---- stage P (16x64 bf16) through per-wave LDS to A-layout ----
#pragma unroll
        for (int nn = 0; nn < 4; ++nn)
#pragma unroll
            for (int r = 0; r < 8; ++r)
                myP[(rb + r) * 64 + nn * 16 + col] = f2bf(S[nn].f[r]);

        ABFrag pa0, pa1;
        {
            const ushort_t* p = myP + col * 64 + half * 8;
            pa0.q[0] = *(const uint4*)(p);
            pa0.q[1] = *(const uint4*)(p + 16);
            pa1.q[0] = *(const uint4*)(p + 32);
            pa1.q[1] = *(const uint4*)(p + 48);
        }

        // ---- O += P V  (V tile rows = hd dims, cols = keys) ----
#pragma unroll
        for (int on = 0; on < 4; ++on) {
            ABFrag vb0, vb1;
            const ushort_t* pv = Vl + (on * 16 + col) * KVP + half * 8;
            vb0.q[0] = *(const uint4*)(pv);       vb0.q[1] = *(const uint4*)(pv + 16);
            vb1.q[0] = *(const uint4*)(pv + 32);  vb1.q[1] = *(const uint4*)(pv + 48);
            O[on].v = wmma_bf16(pa0, vb0, O[on].v);
            O[on].v = wmma_bf16(pa1, vb1, O[on].v);
        }
    }

    // ---- normalize and write attention output bf16 [B][T][H*hd] ----
    const int b = bh >> 4, h = bh & 15;
#pragma unroll
    for (int on = 0; on < 4; ++on) {
#pragma unroll
        for (int r = 0; r < 8; ++r) {
            int t = q0 + rb + r;
            float val = O[on].f[r] / l_r[r];
            Ob[((size_t)(b * T + t)) * 1024 + h * 64 + on * 16 + col] = f2bf(val);
        }
    }
}

// ---------------------------------------------------------------------------
// Host launcher
// ---------------------------------------------------------------------------
extern "C" void kernel_launch(void* const* d_in, const int* in_sizes, int n_in,
                              void* d_out, int out_size, void* d_ws, size_t ws_size,
                              hipStream_t stream) {
    const float* x     = (const float*)d_in[0];
    const float* w_qkv = (const float*)d_in[1];
    const float* b_qkv = (const float*)d_in[2];
    const float* w_out = (const float*)d_in[3];
    const float* b_out = (const float*)d_in[4];
    float* out = (float*)d_out;

    const int B = 2, T = 2048, DM = 1024;
    const size_t MT = (size_t)B * T;                 // 4096 rows

    // workspace carve-up (~88 MB, all 16B-aligned, contiguous — prefetch-safe)
    char* w = (char*)d_ws;
    ushort_t* xb    = (ushort_t*)w;  w += MT * DM * 2;             // 8 MB
    ushort_t* wqkvT = (ushort_t*)w;  w += (size_t)3 * DM * DM * 2; // 6 MB
    ushort_t* woutT = (ushort_t*)w;  w += (size_t)DM * DM * 2;     // 2 MB
    float*    qkv   = (float*)w;     w += MT * 3 * DM * 4;         // 48 MB
    ushort_t* Qb    = (ushort_t*)w;  w += MT * DM * 2;             // 8 MB
    ushort_t* Kb    = (ushort_t*)w;  w += MT * DM * 2;             // 8 MB
    ushort_t* Vt    = (ushort_t*)w;  w += MT * DM * 2;             // 8 MB
    ushort_t* attnb = xb;            // reuse: x_bf16 dead after GEMM1

    f2bf_kernel<<<(unsigned)((MT * DM) / 256), 256, 0, stream>>>(x, xb, (unsigned)(MT * DM));
    transpose_f2bf<<<dim3(3 * DM / 32, DM / 32), dim3(32, 8), 0, stream>>>(w_qkv, wqkvT, DM, 3 * DM);
    transpose_f2bf<<<dim3(DM / 32, DM / 32), dim3(32, 8), 0, stream>>>(w_out, woutT, DM, DM);
    gemm_bf16<<<dim3(3 * DM / GBN, MT / GBM), 256, 0, stream>>>(xb, wqkvT, b_qkv, qkv,
                                                                (int)MT, 3 * DM, DM);
    rope_pack<<<(unsigned)((MT * DM) / 256), 256, 0, stream>>>(qkv, Qb, Kb, Vt);
    attn_kernel<<<dim3(T / 128, B * 16), 256, 0, stream>>>(Qb, Kb, Vt, attnb);
    gemm_bf16<<<dim3(DM / GBN, MT / GBM), 256, 0, stream>>>(attnb, woutT, b_out, out,
                                                            (int)MT, DM, DM);
}
